// LeakyDecisionAccumulator_23210003268376
// MI455X (gfx1250) — matrix operand order repaired and checked by reading, not compiled, not run on hardware
//
#include <hip/hip_runtime.h>
#include <hip/hip_bf16.h>

// LeakyDecisionAccumulator for MI455X (gfx1250, wave32, WMMA).
// Phase 1: fused triple GEMM (We/Wl/Ww) + bias + sigmoid -> workspace [t][b][d]:
//          evidence as f16, gates packed as (l,w) half2 (one b32 per element).
// Phase 2: serial scan over T; Wp^T resident in VGPR B-fragments, belief
//          round-tripped through DOUBLE-BUFFERED LDS f16 (1 barrier/step),
//          WMMA chain split into two accumulators to halve the dep chain.
// Workspace requirement: T*B*D*(2 + 4) bytes = 384 MB.

typedef __attribute__((ext_vector_type(16))) _Float16     v16h;
typedef __attribute__((ext_vector_type(2)))  _Float16     v2h;
typedef __attribute__((ext_vector_type(8)))  float        v8f;
typedef __attribute__((ext_vector_type(8)))  unsigned int v8u;

#define D_ 128
#define T_ 2048
#define B_ 256
#define PITCH (D_ + 8)   // halves; pad to stagger LDS banks

__device__ __forceinline__ float sigmoid_f(float x) {
  return 1.0f / (1.0f + __expf(-x));
}
__device__ __forceinline__ float tanh_f(float x) {
  float t = __expf(-2.0f * fabsf(x));
  float r = (1.0f - t) / (1.0f + t);
  return copysignf(r, x);
}

// A-fragment (16x32 f16) from LDS, row-major [M][PITCH] halves.
// Lane m = lane&15, khalf = lane>>4. VGPRs 0..3: K = kc*32 + khalf*8 + {0..7};
// VGPRs 4..7: K = kc*32 + 16 + khalf*8 + {0..7}. Even-K pairs -> dword loads.
__device__ __forceinline__ v16h load_a_frag(const _Float16* lds, int mrow, int kc, int khalf) {
  const unsigned int* p = (const unsigned int*)lds;
  const int base0 = (mrow * PITCH + kc * 32 + khalf * 8) >> 1;
  const int base1 = (mrow * PITCH + kc * 32 + 16 + khalf * 8) >> 1;
  v8u u;
#pragma unroll
  for (int v = 0; v < 4; ++v) u[v] = p[base0 + v];
#pragma unroll
  for (int v = 0; v < 4; ++v) u[4 + v] = p[base1 + v];
  return __builtin_bit_cast(v16h, u);
}

// B-fragment (32x16 f16) for out = x @ W^T: B(k,n) = W[n][k].
// Lane n = lane&15, khalf = lane>>4: halves 0..7 hold K = kc*32 + khalf*8 + {0..7}, halves 8..15 hold K = kc*32 + 16 + khalf*8 + {0..7}.
__device__ __forceinline__ v16h load_b_frag(const float* __restrict__ W, int n, int kc, int khalf) {
  const float* row = W + n * D_ + kc * 32 + khalf * 8;
  v16h f;
#pragma unroll
  for (int r = 0; r < 16; ++r) f[r] = (_Float16)row[r < 8 ? r : r + 8];
  return f;
}

// ---------------- Phase 1: batched triple projection GEMM ----------------
__global__ __launch_bounds__(256, 1) void proj3_kernel(
    const float* __restrict__ x,
    const float* __restrict__ We, const float* __restrict__ be,
    const float* __restrict__ Wl, const float* __restrict__ bl,
    const float* __restrict__ Ww, const float* __restrict__ bw,
    _Float16* __restrict__ e16, v2h* __restrict__ lw)
{
  __shared__ alignas(16) _Float16 xs[32 * PITCH];
  const int tid   = threadIdx.x;
  const int wave  = tid >> 5;
  const int lane  = tid & 31;
  const int nloc  = lane & 15;
  const int khalf = lane >> 4;
  const int n     = wave * 16 + nloc;
  const int row0  = blockIdx.x * 32;   // flat (b*T + t) row

  // Stage x tile 32x128 (f32 -> f16) into LDS.
#pragma unroll
  for (int i = 0; i < 16; ++i) {
    int idx = tid + i * 256;
    int r = idx >> 7, c = idx & 127;
    xs[r * PITCH + c] = (_Float16)x[(row0 + r) * D_ + c];
  }

  // Resident B fragments for the 3 weight matrices (96 VGPRs).
  v16h bf[3][4];
#pragma unroll
  for (int kc = 0; kc < 4; ++kc) {
    bf[0][kc] = load_b_frag(We, n, kc, khalf);
    bf[1][kc] = load_b_frag(Wl, n, kc, khalf);
    bf[2][kc] = load_b_frag(Ww, n, kc, khalf);
  }
  const float bias0 = be[n], bias1 = bl[n], bias2 = bw[n];

  __syncthreads();

  const v8f zero = {};
  v8f acc[3][2];
#pragma unroll
  for (int m = 0; m < 3; ++m)
#pragma unroll
    for (int s = 0; s < 2; ++s) acc[m][s] = zero;

#pragma unroll
  for (int kc = 0; kc < 4; ++kc) {
    v16h a0 = load_a_frag(xs, nloc,      kc, khalf);
    v16h a1 = load_a_frag(xs, 16 + nloc, kc, khalf);
#pragma unroll
    for (int m = 0; m < 3; ++m) {
      acc[m][0] = __builtin_amdgcn_wmma_f32_16x16x32_f16(
          false, a0, false, bf[m][kc], (short)0, acc[m][0], false, false);
      acc[m][1] = __builtin_amdgcn_wmma_f32_16x16x32_f16(
          false, a1, false, bf[m][kc], (short)0, acc[m][1], false, false);
    }
  }

  // Epilogue: bias (+sigmoid for gates), store workspace in [t][b][d].
#pragma unroll
  for (int s = 0; s < 2; ++s) {
#pragma unroll
    for (int r = 0; r < 8; ++r) {
      int M  = s * 16 + r + 8 * khalf;
      int rg = row0 + M;
      int b  = rg >> 11;        // / T_ (2048)
      int t  = rg & (T_ - 1);
      int addr = (t * B_ + b) * D_ + n;
      e16[addr] = (_Float16)(acc[0][s][r] + bias0);
      v2h g;
      g[0] = (_Float16)sigmoid_f(acc[1][s][r] + bias1);
      g[1] = (_Float16)sigmoid_f(acc[2][s][r] + bias2);
      lw[addr] = g;
    }
  }
}

// ---------------- Phase 2: serial belief scan ----------------
__global__ __launch_bounds__(256, 1) void scan_kernel(
    const float* __restrict__ Wp, const float* __restrict__ bp,
    const _Float16* __restrict__ e16, const v2h* __restrict__ lw,
    float* __restrict__ out)
{
  __shared__ alignas(16) _Float16 bel[2][16 * PITCH];   // double-buffered
  const int tid   = threadIdx.x;
  const int wave  = tid >> 5;
  const int lane  = tid & 31;
  const int nloc  = lane & 15;
  const int khalf = lane >> 4;
  const int n     = wave * 16 + nloc;
  const int b0    = blockIdx.x * 16;

  // Wp^T B-fragments resident for the whole scan (32 VGPRs).
  v16h bf[4];
#pragma unroll
  for (int kc = 0; kc < 4; ++kc) bf[kc] = load_b_frag(Wp, n, kc, khalf);
  const float biasp = bp[n];

  // Only buffer 0 is read at t=0; buffer 1 is written before it is read.
  for (int i = tid; i < 16 * PITCH; i += 256) bel[0][i] = (_Float16)0.0f;

  const v8f zero = {};
  v8f belief = zero;            // this wave's own 16x16 tile, kept in f32
  __syncthreads();

  for (int t = 0; t < T_; ++t) {
    const _Float16* rbuf = bel[t & 1];
    _Float16*       wbuf = bel[(t + 1) & 1];

    // Stream precomputed evidence / gates (overlaps the WMMA chain below).
    float ev[8], lv[8], wv[8];
    const int base = (t * B_ + b0) * D_ + n;
#pragma unroll
    for (int r = 0; r < 8; ++r) {
      int a = base + (r + 8 * khalf) * D_;
      ev[r] = (float)e16[a];
      v2h g = lw[a];
      lv[r] = (float)g[0];
      wv[r] = (float)g[1];
    }

    // belief_prev @ Wp^T: two independent 2-WMMA chains, then combine.
    v8f acc0 = zero, acc1 = zero;
#pragma unroll
    for (int kp = 0; kp < 2; ++kp) {
      v16h a0 = load_a_frag(rbuf, nloc, 2 * kp,     khalf);
      v16h a1 = load_a_frag(rbuf, nloc, 2 * kp + 1, khalf);
      acc0 = __builtin_amdgcn_wmma_f32_16x16x32_f16(
          false, a0, false, bf[2 * kp],     (short)0, acc0, false, false);
      acc1 = __builtin_amdgcn_wmma_f32_16x16x32_f16(
          false, a1, false, bf[2 * kp + 1], (short)0, acc1, false, false);
    }
    v8f acc = acc0 + acc1;

#pragma unroll
    for (int r = 0; r < 8; ++r) {
      float cand = tanh_f(acc[r] + biasp + ev[r]);
      belief[r]  = lv[r] * belief[r] + wv[r] * cand;
    }

    // Emit f32 output and publish f16 belief into the *other* buffer.
#pragma unroll
    for (int r = 0; r < 8; ++r) {
      int M = r + 8 * khalf;
      out[((b0 + M) * T_ + t) * D_ + n] = belief[r];
      wbuf[M * PITCH + n] = (_Float16)belief[r];
    }
    // Single barrier: guarantees (a) wbuf visible for step t+1's reads and
    // (b) everyone is done reading rbuf before step t+1 overwrites it as
    // its wbuf. Safe with one fence.
    __syncthreads();
  }
}

extern "C" void kernel_launch(void* const* d_in, const int* in_sizes, int n_in,
                              void* d_out, int out_size, void* d_ws, size_t ws_size,
                              hipStream_t stream) {
  const float* x  = (const float*)d_in[0];
  const float* We = (const float*)d_in[1];
  const float* be = (const float*)d_in[2];
  const float* Wp = (const float*)d_in[3];
  const float* bp = (const float*)d_in[4];
  const float* Wl = (const float*)d_in[5];
  const float* bl = (const float*)d_in[6];
  const float* Ww = (const float*)d_in[7];
  const float* bw = (const float*)d_in[8];
  float* out = (float*)d_out;

  const size_t N = (size_t)T_ * B_ * D_;
  _Float16* e16 = (_Float16*)d_ws;          // N halves   = 128 MB
  v2h*      lwp = (v2h*)(e16 + N);          // N half2s   = 256 MB

  proj3_kernel<<<(B_ * T_) / 32, 256, 0, stream>>>(x, We, be, Wl, bl, Ww, bw,
                                                   e16, lwp);
  scan_kernel<<<B_ / 16, 256, 0, stream>>>(Wp, bp, e16, lwp, out);
}
